// SelectiveSSM_71244917506114
// MI455X (gfx1250) — matrix operand order repaired and checked by reading, not compiled, not run on hardware
//
#include <hip/hip_runtime.h>
#include <hip/hip_bf16.h>
#include <math.h>

typedef __attribute__((ext_vector_type(16))) _Float16 v16h;
typedef __attribute__((ext_vector_type(8)))  _Float16 v8h;
typedef __attribute__((ext_vector_type(8)))  float    v8f;

#define BATCH   2
#define SEQ     2048
#define DMODEL  512
#define DINNER  1024
#define DSTATE  16
#define DCONV   4
#define BL      (BATCH*SEQ)      /* 4096 rows */
#define EPS     1e-6f

// ---------------------------------------------------------------------------
// Elementwise helpers
// ---------------------------------------------------------------------------
__device__ __forceinline__ float silu_f(float v) { return v / (1.f + __expf(-v)); }

// f32 -> f16 copy
__global__ void k_f32_to_f16(const float* __restrict__ src, _Float16* __restrict__ dst, int n) {
    int i = blockIdx.x * 256 + threadIdx.x;
    if (i < n) dst[i] = (_Float16)src[i];
}

// W (K x N, f32 row-major) -> Wt (N x K, f16 row-major)
__global__ void k_transpose_f16(const float* __restrict__ W, _Float16* __restrict__ Wt, int K, int N) {
    int i = blockIdx.x * 256 + threadIdx.x;
    if (i >= K * N) return;
    int k = i / N, n = i % N;
    Wt[(size_t)n * K + k] = (_Float16)W[i];
}

// Causal depthwise conv (width 4, left pad 3) + bias + SiLU over x_proj = xz[:, :DINNER]
__global__ void k_conv_silu(const float* __restrict__ xz, const float* __restrict__ w,
                            const float* __restrict__ bias,
                            float* __restrict__ xc, _Float16* __restrict__ xch) {
    int i = blockIdx.x * 256 + threadIdx.x;          // over BL*DINNER
    if (i >= BL * DINNER) return;
    int ch  = i & (DINNER - 1);
    int row = i >> 10;                               // b*SEQ + l
    int l   = row & (SEQ - 1);
    float acc = bias[ch];
#pragma unroll
    for (int j = 0; j < DCONV; ++j) {
        int ll = l - (DCONV - 1) + j;
        if (ll >= 0)
            acc += w[ch * DCONV + j] * xz[(size_t)(row - (DCONV - 1) + j) * (2 * DINNER) + ch];
    }
    float s = silu_f(acc);
    xc[i]  = s;
    xch[i] = (_Float16)s;
}

// y_gated = (y + D*xc) * silu(z), emitted as f16 for the output GEMM
__global__ void k_gate(const float* __restrict__ y, const float* __restrict__ xc,
                       const float* __restrict__ xz, const float* __restrict__ Dv,
                       _Float16* __restrict__ ygh) {
    int i = blockIdx.x * 256 + threadIdx.x;          // over BL*DINNER
    if (i >= BL * DINNER) return;
    int ch  = i & (DINNER - 1);
    int row = i >> 10;
    float z = xz[(size_t)row * (2 * DINNER) + DINNER + ch];
    float v = (y[i] + Dv[ch] * xc[i]) * silu_f(z);
    ygh[i] = (_Float16)v;
}

// ---------------------------------------------------------------------------
// Chunked selective scan: one lane per (batch, channel, state); 16-lane group
// per channel reduced with shfl_xor. Exactly mirrors the reference chunk math.
// ---------------------------------------------------------------------------
__global__ void k_scan(const float* __restrict__ dt, const float* __restrict__ xc,
                       const float* __restrict__ bc, const float* __restrict__ A_log,
                       float* __restrict__ y) {
    int t = blockIdx.x * 256 + threadIdx.x;          // 0 .. BATCH*DINNER*DSTATE-1
    int s = t & 15;
    int i = (t >> 4) & (DINNER - 1);
    int b = t >> 14;
    float A = -__expf(A_log[i * DSTATE + s]);
    const float* dt_p = dt + (size_t)b * SEQ * DINNER + i;
    const float* xc_p = xc + (size_t)b * SEQ * DINNER + i;
    const float* bc_p = bc + (size_t)b * SEQ * 32;
    float*       y_p  = y  + (size_t)b * SEQ * DINNER + i;

    float h0 = 0.f, cum_dA = 1.f, ratio = 0.f, h = 0.f;
    for (int l = 0; l < SEQ; ++l) {
        float dtv = dt_p[(size_t)l * DINNER];
        float xcv = xc_p[(size_t)l * DINNER];
        float Bv  = bc_p[l * 32 + s];
        float Cv  = bc_p[l * 32 + 16 + s];
        float dA  = __expf(dtv * A);
        cum_dA *= dA;
        ratio  += (dtv * Bv * xcv) / (cum_dA + EPS);
        h = cum_dA * (h0 + ratio);
        float contrib = h * Cv;
        contrib += __shfl_xor(contrib, 1);
        contrib += __shfl_xor(contrib, 2);
        contrib += __shfl_xor(contrib, 4);
        contrib += __shfl_xor(contrib, 8);
        if (s == 0) y_p[(size_t)l * DINNER] = contrib;
        if ((l & 31) == 31) { h0 = h; cum_dA = 1.f; ratio = 0.f; }
    }
}

// ---------------------------------------------------------------------------
// WMMA GEMM: C(MxN,f32) = A(MxK,f16 row-major) * Wt(NxK,f16 row-major)^T
// Per-wave 16x(NT*16) strip. Unroll-by-2 rotating double buffer (no rotation
// copies). Addressing: uniform SGPR base + 32-bit per-lane BYTE offsets so the
// backend emits saddr-form global_load_b128 with immediate deltas (LOADcnt
// only, no 64-bit VGPR address math). Requires K % 64 == 0.
// Fragments use contiguous k = 16*(lane>>4) + s on both A and B (same order on both).
// ---------------------------------------------------------------------------
__device__ __forceinline__ v16h ld_a(const _Float16* __restrict__ A, unsigned ob) {
    // ob (bytes) already includes (lane>>4)*32B: lanes 0-15 K=k0..k0+15, 16-31 +16
    const char* p = (const char*)A + ob;
    v8h lo = *(const v8h*)p;
    v8h hi = *(const v8h*)(p + 16);
    return __builtin_shufflevector(lo, hi, 0, 1, 2, 3, 4, 5, 6, 7,
                                   8, 9, 10, 11, 12, 13, 14, 15);
}

__device__ __forceinline__ v16h ld_b(const _Float16* __restrict__ Wt, unsigned ob) {
    // ob (bytes) already includes (lane>>4)*32B: lanes 0-15 K=k0..k0+15, 16-31 +16
    return *(const v16h*)((const char*)Wt + ob);
}

__device__ __forceinline__ v8f wmma_f16(v16h a, v16h b, v8f c) {
    return __builtin_amdgcn_wmma_f32_16x16x32_f16(false, a, false, b, (short)0, c,
                                                  false, false);
}

template <int EPI>  // 0 = none, 1 = +bias then softplus
__device__ __forceinline__ void store_c(float* __restrict__ C, int ldc, int m0, int n0,
                                        int lane, v8f c, const float* __restrict__ bias) {
    int n  = n0 + (lane & 15);
    int mh = (lane >> 4) * 8;
#pragma unroll
    for (int r = 0; r < 8; ++r) {
        float v = c[r];
        if (EPI == 1) {
            v += bias[n];
            v = (v > 20.f) ? v : log1pf(__expf(v));
        }
        C[(size_t)(m0 + mh + r) * ldc + n] = v;
    }
}

template <int EPI, int NT>  // NT = number of 16-wide N sub-tiles per wave
__global__ __launch_bounds__(128) void k_gemm_wmma(const _Float16* __restrict__ A,
                                                   const _Float16* __restrict__ Wt,
                                                   float* __restrict__ C,
                                                   const float* __restrict__ bias,
                                                   int M, int N, int K, int ldc) {
    const int lane = threadIdx.x & 31;
    const int wave = threadIdx.x >> 5;               // 4 waves per block
    const int m0 = (blockIdx.y * 4 + wave) * 16;
    const int n0 = blockIdx.x * (NT * 16);
    if (m0 >= M || n0 >= N) return;                  // wave-uniform: EXEC stays full

    const int lo16 = lane & 15, hi = lane >> 4;
    // Per-lane byte offsets from the uniform base pointers.
    unsigned oa = ((unsigned)(m0 + lo16) * (unsigned)K + (unsigned)(hi * 16)) * 2u;
    unsigned ob[NT];
#pragma unroll
    for (int t = 0; t < NT; ++t)
        ob[t] = ((unsigned)(n0 + t * 16 + lo16) * (unsigned)K + (unsigned)(hi * 16)) * 2u;

    v8f acc[NT];
#pragma unroll
    for (int t = 0; t < NT; ++t) acc[t] = v8f{};

    const int S = K >> 5;                            // #k-steps, even (K % 64 == 0)

    // set0 <- step 0
    v16h a0 = ld_a(A, oa);
    v16h b0[NT];
#pragma unroll
    for (int t = 0; t < NT; ++t) b0[t] = ld_b(Wt, ob[t]);

    for (int i = 0; i < (S >> 1) - 1; ++i) {
        // set1 <- odd step, consume set0
        v16h a1 = ld_a(A, oa + 64);
        v16h b1[NT];
#pragma unroll
        for (int t = 0; t < NT; ++t) b1[t] = ld_b(Wt, ob[t] + 64);
#pragma unroll
        for (int t = 0; t < NT; ++t) acc[t] = wmma_f16(a0, b0[t], acc[t]);
        // set0 <- even step, consume set1
        a0 = ld_a(A, oa + 128);
#pragma unroll
        for (int t = 0; t < NT; ++t) b0[t] = ld_b(Wt, ob[t] + 128);
#pragma unroll
        for (int t = 0; t < NT; ++t) acc[t] = wmma_f16(a1, b1[t], acc[t]);
        oa += 128;
#pragma unroll
        for (int t = 0; t < NT; ++t) ob[t] += 128;
    }
    // tail: set1 <- last step
    {
        v16h a1 = ld_a(A, oa + 64);
        v16h b1[NT];
#pragma unroll
        for (int t = 0; t < NT; ++t) b1[t] = ld_b(Wt, ob[t] + 64);
#pragma unroll
        for (int t = 0; t < NT; ++t) acc[t] = wmma_f16(a0, b0[t], acc[t]);
#pragma unroll
        for (int t = 0; t < NT; ++t) acc[t] = wmma_f16(a1, b1[t], acc[t]);
    }

#pragma unroll
    for (int t = 0; t < NT; ++t)
        store_c<EPI>(C, ldc, m0, n0 + t * 16, lane, acc[t], bias);
}

// ---------------------------------------------------------------------------
// Orchestration
// ---------------------------------------------------------------------------
extern "C" void kernel_launch(void* const* d_in, const int* in_sizes, int n_in,
                              void* d_out, int out_size, void* d_ws, size_t ws_size,
                              hipStream_t stream) {
    const float* x         = (const float*)d_in[0];
    const float* in_proj_w = (const float*)d_in[1];   // (512, 2048)
    const float* conv_w    = (const float*)d_in[2];   // (1024, 1, 4)
    const float* conv_b    = (const float*)d_in[3];
    const float* dt_w      = (const float*)d_in[4];   // (1024, 1024)
    const float* dt_b      = (const float*)d_in[5];
    const float* B_w       = (const float*)d_in[6];   // (1024, 16)
    const float* C_w       = (const float*)d_in[7];   // (1024, 16)
    const float* A_log     = (const float*)d_in[8];   // (1024, 16)
    const float* Dv        = (const float*)d_in[9];
    const float* out_w     = (const float*)d_in[10];  // (1024, 512)
    float* out = (float*)d_out;

    char* ws = (char*)d_ws;
    size_t off = 0;
    auto alloc = [&](size_t bytes) -> char* {
        char* p = ws + off;
        off += (bytes + 255) & ~(size_t)255;
        return p;
    };
    float*    xz     = (float*)alloc((size_t)BL * 2 * DINNER * 4);   // 32 MB
    float*    xc     = (float*)alloc((size_t)BL * DINNER * 4);       // 16 MB
    _Float16* xch    = (_Float16*)alloc((size_t)BL * DINNER * 2);    //  8 MB
    float*    dt     = (float*)alloc((size_t)BL * DINNER * 4);       // 16 MB
    float*    bc     = (float*)alloc((size_t)BL * 32 * 4);           // B|C per row
    float*    y      = (float*)alloc((size_t)BL * DINNER * 4);       // 16 MB
    _Float16* ygh    = (_Float16*)alloc((size_t)BL * DINNER * 2);
    _Float16* xh     = (_Float16*)alloc((size_t)BL * DMODEL * 2);
    _Float16* w_inp  = (_Float16*)alloc((size_t)2 * DINNER * DMODEL * 2);  // (2048,512)
    _Float16* w_dt   = (_Float16*)alloc((size_t)DINNER * DINNER * 2);      // (1024,1024)
    _Float16* w_bc   = (_Float16*)alloc((size_t)32 * DINNER * 2);          // (32,1024)
    _Float16* w_out  = (_Float16*)alloc((size_t)DMODEL * DINNER * 2);      // (512,1024)

    // Precision conversions / weight transposes (one pass, small).
    k_f32_to_f16<<<(BL * DMODEL + 255) / 256, 256, 0, stream>>>(x, xh, BL * DMODEL);
    k_transpose_f16<<<(DMODEL * 2 * DINNER + 255) / 256, 256, 0, stream>>>(in_proj_w, w_inp, DMODEL, 2 * DINNER);
    k_transpose_f16<<<(DINNER * DINNER + 255) / 256, 256, 0, stream>>>(dt_w, w_dt, DINNER, DINNER);
    k_transpose_f16<<<(DINNER * DSTATE + 255) / 256, 256, 0, stream>>>(B_w, w_bc, DINNER, DSTATE);
    k_transpose_f16<<<(DINNER * DSTATE + 255) / 256, 256, 0, stream>>>(C_w, w_bc + (size_t)DSTATE * DINNER, DINNER, DSTATE);
    k_transpose_f16<<<(DINNER * DMODEL + 255) / 256, 256, 0, stream>>>(out_w, w_out, DINNER, DMODEL);

    // 1) xz = x @ in_proj_w            (4096 x 2048, K=512)
    dim3 g1(2 * DINNER / 64, BL / 64);
    k_gemm_wmma<0, 4><<<g1, 128, 0, stream>>>(xh, w_inp, xz, nullptr, BL, 2 * DINNER, DMODEL, 2 * DINNER);

    // 2) causal depthwise conv + SiLU -> xc (f32 + f16)
    k_conv_silu<<<(BL * DINNER + 255) / 256, 256, 0, stream>>>(xz, conv_w, conv_b, xc, xch);

    // 3) dt = softplus(xc @ dt_w + dt_b)   (4096 x 1024, K=1024)
    dim3 g2(DINNER / 64, BL / 64);
    k_gemm_wmma<1, 4><<<g2, 128, 0, stream>>>(xch, w_dt, dt, dt_b, BL, DINNER, DINNER, DINNER);

    // 4) [B|C] = xc @ [B_w|C_w]            (4096 x 32, K=1024), N = 2 sub-tiles
    dim3 g3(1, BL / 64);
    k_gemm_wmma<0, 2><<<g3, 128, 0, stream>>>(xch, w_bc, bc, nullptr, BL, 32, DINNER, 32);

    // 5) chunked selective scan -> y
    k_scan<<<(BATCH * DINNER * DSTATE) / 256, 256, 0, stream>>>(dt, xc, bc, A_log, y);

    // 6) gating: yg = (y + D*xc) * silu(z)
    k_gate<<<(BL * DINNER + 255) / 256, 256, 0, stream>>>(y, xc, xz, Dv, ygh);

    // 7) out = yg @ out_proj_w             (4096 x 512, K=1024)
    dim3 g4(DMODEL / 64, BL / 64);
    k_gemm_wmma<0, 4><<<g4, 128, 0, stream>>>(ygh, w_out, out, nullptr, BL, DMODEL, DINNER, DMODEL);
}
